// SelfAttention_8143257994068
// MI455X (gfx1250) — matrix operand order repaired and checked by reading, not compiled, not run on hardware
//
#include <hip/hip_runtime.h>

#define D_MODEL 1024
#define NHEADS  16
#define HDIM    64
#define BATCH   4
#define SEQ     2048
#define MTOT    (BATCH * SEQ)   // 8192

typedef __attribute__((ext_vector_type(16))) __bf16 v16bf;
typedef __attribute__((ext_vector_type(8)))  float  v8f;
typedef unsigned int v4u __attribute__((ext_vector_type(4)));
typedef int          v4i __attribute__((ext_vector_type(4)));
typedef int          v8i __attribute__((ext_vector_type(8)));

#if __has_builtin(__builtin_amdgcn_tensor_load_to_lds)
#define HAVE_TDM 1
#else
#define HAVE_TDM 0
#endif

union Frag {
    v16bf bf;
    unsigned int u[8];
};

__device__ __forceinline__ unsigned short f2bf(float f) {
    unsigned int u = __float_as_uint(f);
    unsigned int r = 0x7FFFu + ((u >> 16) & 1u);  // round-to-nearest-even
    return (unsigned short)((u + r) >> 16);
}

#if __has_builtin(__builtin_amdgcn_cvt_pk_bf16_f32)
__device__ __forceinline__ unsigned int pack2bf(float x, float y) {
    auto r = __builtin_amdgcn_cvt_pk_bf16_f32(x, y);
    return __builtin_bit_cast(unsigned int, r);
}
#else
__device__ __forceinline__ unsigned int pack2bf(float x, float y) {
    return (unsigned)f2bf(x) | ((unsigned)f2bf(y) << 16);
}
#endif

__device__ __forceinline__ unsigned int lds_offset(const void* p) {
    // generic LDS address: low 32 bits == byte offset into LDS
    return (unsigned int)(uintptr_t)p;
}

#if HAVE_TDM
// TDM 2D tile load: tile1 rows x tile0 bf16 elements (row = 128B when
// tile0 == 64), memory row stride = stride0 elements, LDS pitch = 144B
// (128B data + 16B pad -> pad_interval 128B, pad_amount 4 DWORDs).
__device__ __forceinline__ void tdm_load_2d(const void* gsrc, unsigned int ldsOff,
                                            unsigned td0, unsigned td1,
                                            unsigned tile0, unsigned tile1,
                                            unsigned stride0)
{
    unsigned long long ga = (unsigned long long)(uintptr_t)gsrc;
    v4u g0;
    g0.x = 0x1u;                                             // count=1, user mode
    g0.y = ldsOff;                                           // lds_addr
    g0.z = (unsigned int)(ga & 0xFFFFFFFFu);                 // global_addr[31:0]
    g0.w = (unsigned int)((ga >> 32) & 0x1FFFFFFu)           // global_addr[56:32]
         | (2u << 30);                                       // type = 2 (image)
    v8i g1;
    g1[0] = (int)((1u << 16)        // data_size = 1 -> 2 bytes
                | (1u << 20)        // pad_enable
                | (4u << 22)        // pad_interval: 8<<4 = 128 bytes
                | (3u << 25));      // pad_amount: 4 DWORDs = 16 bytes
    g1[1] = (int)(td0 << 16);                        // tensor_dim0[15:0]
    g1[2] = (int)((td0 >> 16) | (td1 << 16));        // dim0 hi | dim1 lo
    g1[3] = (int)((td1 >> 16) | (tile0 << 16));      // dim1 hi | tile_dim0
    g1[4] = (int)tile1;                              // tile_dim1 (tile_dim2=0)
    g1[5] = (int)stride0;                            // tensor_dim0_stride
    g1[6] = 0;
    g1[7] = 0;
    v4i gz = {0, 0, 0, 0};
#if __clang_major__ >= 23
    v8i gz8 = {0, 0, 0, 0, 0, 0, 0, 0};
    __builtin_amdgcn_tensor_load_to_lds(g0, g1, gz, gz, gz8, 0);
#else
    __builtin_amdgcn_tensor_load_to_lds(g0, g1, gz, gz, 0);
#endif
}
#endif

// ---------------------------------------------------------------------------
// fp32 -> bf16 bulk conversion (one pass; all GEMMs then consume pure bf16)
// ---------------------------------------------------------------------------
__global__ __launch_bounds__(256)
void cvt_bf16(const float* __restrict__ src, unsigned short* __restrict__ dst, int n)
{
    int i = (blockIdx.x * 256 + threadIdx.x) * 8;
    if (i + 7 < n) {
        float4 v0 = *(const float4*)(src + i);
        float4 v1 = *(const float4*)(src + i + 4);
        uint4 r;
        r.x = pack2bf(v0.x, v0.y);
        r.y = pack2bf(v0.z, v0.w);
        r.z = pack2bf(v1.x, v1.y);
        r.w = pack2bf(v1.z, v1.w);
        *(uint4*)(dst + i) = r;
    }
}

// ---------------------------------------------------------------------------
// bf16 GEMM: out[m][n] = sum_k A[m][k] * W[n][k] + bias[n]
// A bf16 [8192 x 1024], W bf16 [1024 x 1024].
// 128(M) x 64(N) block, BK=64, 8 waves; TDM double-buffered LDS tiles.
// MODE 0: fp32 out row-major [M x D]
// MODE 1: bf16 out [B,H,S,Hd]          (Q, K)
// MODE 2: bf16 out [B,H,Hd,S] (V^T for contiguous P*V fragment loads)
// ---------------------------------------------------------------------------
template<int MODE>
__global__ __launch_bounds__(256)
void gemm_bf16(const unsigned short* __restrict__ A,
               const unsigned short* __restrict__ W,
               const float* __restrict__ bias,
               unsigned short* __restrict__ outB,
               float* __restrict__ outF)
{
    const int LP = 72;   // bf16 row pitch (144B = 128B data + 16B pad)
    __shared__ __align__(16) unsigned short As[2][128 * 72];
    __shared__ __align__(16) unsigned short Bs[2][64 * 72];

    const int tid  = threadIdx.x;
    const int lane = tid & 31;
    const int wid  = tid >> 5;
    const int m0   = blockIdx.x * 128;
    const int n0   = blockIdx.y * 64;

    v8f acc[4] = {};

#if HAVE_TDM
    if (tid < 32) {   // wave 0 issues DMA (EXEC ignored by TDM)
        tdm_load_2d(A + (size_t)m0 * D_MODEL, lds_offset(&As[0][0]),
                    D_MODEL, MTOT, 64, 128, D_MODEL);
        tdm_load_2d(W + (size_t)n0 * D_MODEL, lds_offset(&Bs[0][0]),
                    D_MODEL, D_MODEL, 64, 64, D_MODEL);
    }
#endif

    for (int kt = 0; kt < D_MODEL; kt += 64) {
        const int cur = (kt >> 6) & 1;
        __syncthreads();   // compute on buffer cur^1 from previous iter done
#if HAVE_TDM
        if (tid < 32) {
            if (kt + 64 < D_MODEL) {
                tdm_load_2d(A + (size_t)m0 * D_MODEL + kt + 64,
                            lds_offset(&As[cur ^ 1][0]),
                            D_MODEL, MTOT, 64, 128, D_MODEL);
                tdm_load_2d(W + (size_t)n0 * D_MODEL + kt + 64,
                            lds_offset(&Bs[cur ^ 1][0]),
                            D_MODEL, D_MODEL, 64, 64, D_MODEL);
                __builtin_amdgcn_s_wait_tensorcnt(2);   // tile kt has landed
            } else {
                __builtin_amdgcn_s_wait_tensorcnt(0);
            }
        }
#else
        #pragma unroll
        for (int j = 0; j < 4; ++j) {
            int f = tid * 4 + j, row = f >> 3, c8 = (f & 7) * 8;
            *(uint4*)&As[cur][row * LP + c8] =
                *(const uint4*)(A + (size_t)(m0 + row) * D_MODEL + kt + c8);
        }
        #pragma unroll
        for (int j = 0; j < 2; ++j) {
            int f = tid * 2 + j, row = f >> 3, c8 = (f & 7) * 8;
            *(uint4*)&Bs[cur][row * LP + c8] =
                *(const uint4*)(W + (size_t)(n0 + row) * D_MODEL + kt + c8);
        }
#endif
        __syncthreads();

        #pragma unroll
        for (int ks = 0; ks < 2; ++ks) {
            Frag a, b[4];
            int ar = wid * 16 + (lane & 15);
            int kh = ks * 32 + (lane >> 4) * 8;
            #pragma unroll
            for (int j = 0; j < 4; ++j) {
                a.u[j]     = *(const unsigned int*)&As[cur][ar * LP + kh + 2 * j];
                a.u[j + 4] = *(const unsigned int*)&As[cur][ar * LP + 16 + kh + 2 * j];
            }
            int bn = lane & 15;
            int kb = ks * 32 + (lane >> 4) * 8;
            #pragma unroll
            for (int nb = 0; nb < 4; ++nb)
                #pragma unroll
                for (int j = 0; j < 8; ++j)
                    b[nb].u[j] = *(const unsigned int*)&Bs[cur][(nb * 16 + bn) * LP + kb + 2 * j + (j >> 2) * 8];
            #pragma unroll
            for (int nb = 0; nb < 4; ++nb)
                acc[nb] = __builtin_amdgcn_wmma_f32_16x16x32_bf16(
                    false, a.bf, false, b[nb].bf, (short)0, acc[nb], false, false);
        }
    }

    // Epilogue: C layout VGPR i -> M = i (+8 upper lane half), N = lane&15
    int mbase = m0 + wid * 16 + ((lane >> 4) * 8);
    #pragma unroll
    for (int nb = 0; nb < 4; ++nb) {
        int n  = n0 + nb * 16 + (lane & 15);
        float bv = bias[n];
        if constexpr (MODE == 1) {
            int h = n >> 6, d = n & 63;
            #pragma unroll
            for (int i = 0; i < 8; ++i) {
                int m = mbase + i;
                int b = m >> 11, s = m & 2047;
                size_t idx = (((size_t)(b * NHEADS + h)) * SEQ + s) * HDIM + d;
                outB[idx] = f2bf(acc[nb][i] + bv);
            }
        } else if constexpr (MODE == 2) {
            int h = n >> 6, d = n & 63;
            #pragma unroll
            for (int i = 0; i < 8; ++i) {
                int m = mbase + i;
                int b = m >> 11, s = m & 2047;
                size_t idx = (((size_t)(b * NHEADS + h)) * HDIM + d) * SEQ + s;
                outB[idx] = f2bf(acc[nb][i] + bv);
            }
        } else {
            #pragma unroll
            for (int i = 0; i < 8; ++i)
                outF[(size_t)(mbase + i) * D_MODEL + n] = acc[nb][i] + bv;
        }
    }
}

// ---------------------------------------------------------------------------
// Flash attention: one workgroup per (b*H+h, 64-row query block).
// K tiles [s][d] and V^T tiles [d][s] streamed by the TDM into
// double-buffered LDS; all WMMA fragments are contiguous LDS reads.
// ---------------------------------------------------------------------------
__global__ __launch_bounds__(256)
void flash_attn(const unsigned short* __restrict__ Q,
                const unsigned short* __restrict__ K,
                const unsigned short* __restrict__ Vt,
                unsigned short* __restrict__ O)
{
    const int LQ = 72;
    const int LS = 68;
    __shared__ __align__(16) unsigned short Qs[64 * 72];
    __shared__ __align__(16) unsigned short Ks[2][64 * 72];
    __shared__ __align__(16) unsigned short Vs[2][64 * 72];   // [d][s] tile
    __shared__ __align__(16) unsigned short Ps[64 * 72];
    __shared__ __align__(16) float Sc[64 * 68];
    __shared__ float rowM[64], rowL[64], rowC[64];

    const int tid  = threadIdx.x;
    const int lane = tid & 31;
    const int wid  = tid >> 5;
    const int blk  = blockIdx.x;
    const int qb   = blk & 31;
    const int bh   = blk >> 5;
    const size_t headOff = (size_t)bh * SEQ * HDIM;
    const float scale = 0.125f;      // 1/sqrt(64)

    const unsigned short* kbase = K  + headOff;   // [s][d], row stride HDIM
    const unsigned short* vbase = Vt + headOff;   // [d][s], row stride SEQ

    // Load Q tile (64x64 bf16 = 512 uint4, 2 per thread)
    {
        const unsigned short* src = Q + headOff + (size_t)qb * 64 * HDIM;
        #pragma unroll
        for (int j = 0; j < 2; ++j) {
            int f = tid * 2 + j, row = f >> 3, c8 = (f & 7) * 8;
            *(uint4*)&Qs[row * LQ + c8] = *(const uint4*)(src + row * HDIM + c8);
        }
    }
    if (tid < 64) { rowM[tid] = -1e30f; rowL[tid] = 0.0f; }

#if HAVE_TDM
    if (tid < 32) {
        tdm_load_2d(kbase, lds_offset(&Ks[0][0]), HDIM, SEQ, 64, 64, HDIM);
        tdm_load_2d(vbase, lds_offset(&Vs[0][0]), SEQ, HDIM, 64, 64, SEQ);
    }
#endif

    v8f o0 = {}, o1 = {};
    const int mb0 = (wid * 2) >> 2,     db0 = (wid * 2) & 3;
    const int mb1 = (wid * 2 + 1) >> 2, db1 = (wid * 2 + 1) & 3;

    for (int kb = 0; kb < SEQ / 64; ++kb) {
        const int cur = kb & 1;
        __syncthreads();
#if HAVE_TDM
        if (tid < 32) {
            if (kb + 1 < SEQ / 64) {
                tdm_load_2d(kbase + (size_t)(kb + 1) * 64 * HDIM,
                            lds_offset(&Ks[cur ^ 1][0]), HDIM, SEQ, 64, 64, HDIM);
                tdm_load_2d(vbase + (size_t)(kb + 1) * 64,
                            lds_offset(&Vs[cur ^ 1][0]), SEQ, HDIM, 64, 64, SEQ);
                __builtin_amdgcn_s_wait_tensorcnt(2);
            } else {
                __builtin_amdgcn_s_wait_tensorcnt(0);
            }
        }
#else
        #pragma unroll
        for (int j = 0; j < 2; ++j) {
            int f = tid * 2 + j, row = f >> 3, c8 = (f & 7) * 8;
            *(uint4*)&Ks[cur][row * LQ + c8] =
                *(const uint4*)(kbase + (size_t)kb * 64 * HDIM + row * HDIM + c8);
            *(uint4*)&Vs[cur][row * LQ + c8] =
                *(const uint4*)(vbase + (size_t)row * SEQ + kb * 64 + c8);
        }
#endif
        __syncthreads();

        // ---- S = Q * K^T (scaled); each wave 2 of 16 subtiles ----
        #pragma unroll
        for (int u = 0; u < 2; ++u) {
            int t  = wid * 2 + u;
            int mb = t >> 2, nb = t & 3;
            v8f s = {};
            #pragma unroll
            for (int ks = 0; ks < 2; ++ks) {
                Frag a, b;
                int ar = mb * 16 + (lane & 15);
                int kh = (lane >> 4) * 8;
                #pragma unroll
                for (int j = 0; j < 4; ++j) {
                    a.u[j]     = *(const unsigned int*)&Qs[ar * LQ + ks * 32 + kh + 2 * j];
                    a.u[j + 4] = *(const unsigned int*)&Qs[ar * LQ + ks * 32 + 16 + kh + 2 * j];
                }
                int bn  = nb * 16 + (lane & 15);
                int kbs = (lane >> 4) * 8;
                #pragma unroll
                for (int j = 0; j < 8; ++j)
                    b.u[j] = *(const unsigned int*)&Ks[cur][bn * LQ + ks * 32 + kbs + 2 * j + (j >> 2) * 8];
                s = __builtin_amdgcn_wmma_f32_16x16x32_bf16(
                    false, a.bf, false, b.bf, (short)0, s, false, false);
            }
            int rbase = mb * 16 + ((lane >> 4) * 8);
            int col   = nb * 16 + (lane & 15);
            #pragma unroll
            for (int i = 0; i < 8; ++i)
                Sc[(rbase + i) * LS + col] = s[i] * scale;
        }
        __syncthreads();

        // ---- online softmax: thread t < 64 owns one row ----
        if (tid < 64) {
            float mOld = rowM[tid];
            float mNew = mOld;
            const float* sr = &Sc[tid * LS];
            #pragma unroll 8
            for (int c = 0; c < 64; ++c) mNew = fmaxf(mNew, sr[c]);
            float corr = __expf(mOld - mNew);
            float lsum = 0.0f;
            unsigned int* pr = (unsigned int*)&Ps[tid * LQ];
            #pragma unroll 8
            for (int c = 0; c < 64; c += 2) {
                float p0 = __expf(sr[c]     - mNew);
                float p1 = __expf(sr[c + 1] - mNew);
                lsum += p0 + p1;
                pr[c >> 1] = pack2bf(p0, p1);
            }
            rowM[tid] = mNew;
            rowL[tid] = rowL[tid] * corr + lsum;
            rowC[tid] = corr;
        }
        __syncthreads();

        // ---- rescale O accumulators, then O += P * V ----
        {
            int r0 = mb0 * 16 + ((lane >> 4) * 8);
            int r1 = mb1 * 16 + ((lane >> 4) * 8);
            #pragma unroll
            for (int i = 0; i < 8; ++i) {
                o0[i] *= rowC[r0 + i];
                o1[i] *= rowC[r1 + i];
            }
        }
        #pragma unroll
        for (int u = 0; u < 2; ++u) {
            int mb = u ? mb1 : mb0;
            int db = u ? db1 : db0;
            v8f acc = u ? o1 : o0;
            #pragma unroll
            for (int ns = 0; ns < 2; ++ns) {
                Frag a, b;
                int ar = mb * 16 + (lane & 15);
                int kh = (lane >> 4) * 8;
                #pragma unroll
                for (int j = 0; j < 4; ++j) {
                    a.u[j]     = *(const unsigned int*)&Ps[ar * LQ + ns * 32 + kh + 2 * j];
                    a.u[j + 4] = *(const unsigned int*)&Ps[ar * LQ + ns * 32 + 16 + kh + 2 * j];
                }
                // V^T tile: lane's output column d is an LDS row -> contiguous
                int bd  = db * 16 + (lane & 15);
                int kbs = ns * 32 + (lane >> 4) * 8;
                #pragma unroll
                for (int j = 0; j < 8; ++j)
                    b.u[j] = *(const unsigned int*)&Vs[cur][bd * LQ + kbs + 2 * j + (j >> 2) * 8];
                acc = __builtin_amdgcn_wmma_f32_16x16x32_bf16(
                    false, a.bf, false, b.bf, (short)0, acc, false, false);
            }
            if (u) o1 = acc; else o0 = acc;
        }
    }
    __syncthreads();

    // ---- finalize: O /= l ; write bf16 [B,S,D] ----
    const int b = bh >> 4;
    const int h = bh & 15;
    #pragma unroll
    for (int u = 0; u < 2; ++u) {
        int mb = u ? mb1 : mb0;
        int db = u ? db1 : db0;
        v8f acc = u ? o1 : o0;
        int rbase = mb * 16 + ((lane >> 4) * 8);
        int d = db * 16 + (lane & 15);
        #pragma unroll
        for (int i = 0; i < 8; ++i) {
            int r = rbase + i;
            float inv = 1.0f / rowL[r];
            int sg = qb * 64 + r;
            size_t idx = ((size_t)(b * SEQ + sg)) * D_MODEL + h * HDIM + d;
            O[idx] = f2bf(acc[i] * inv);
        }
    }
}

// ---------------------------------------------------------------------------
extern "C" void kernel_launch(void* const* d_in, const int* in_sizes, int n_in,
                              void* d_out, int out_size, void* d_ws, size_t ws_size,
                              hipStream_t stream)
{
    const float* x  = (const float*)d_in[0];
    const float* Wq = (const float*)d_in[1];
    const float* bq = (const float*)d_in[2];
    const float* Wk = (const float*)d_in[3];
    const float* bk = (const float*)d_in[4];
    const float* Wv = (const float*)d_in[5];
    const float* bv = (const float*)d_in[6];
    const float* Wo = (const float*)d_in[7];
    const float* bo = (const float*)d_in[8];
    float* out = (float*)d_out;

    const size_t elems  = (size_t)MTOT * D_MODEL;     // 8M
    const size_t welems = (size_t)D_MODEL * D_MODEL;  // 1M
    unsigned short* xb  = (unsigned short*)d_ws;
    unsigned short* wqb = xb  + elems;
    unsigned short* wkb = wqb + welems;
    unsigned short* wvb = wkb + welems;
    unsigned short* wob = wvb + welems;
    unsigned short* q   = wob + welems;
    unsigned short* k   = q   + elems;
    unsigned short* vT  = k   + elems;
    unsigned short* ao  = vT  + elems;                // total 44M ushorts = 88MB

    // one-shot fp32 -> bf16 conversion of activations and weights
    cvt_bf16<<<(int)(elems  / 8 / 256), 256, 0, stream>>>(x,  xb,  (int)elems);
    cvt_bf16<<<(int)(welems / 8 / 256), 256, 0, stream>>>(Wq, wqb, (int)welems);
    cvt_bf16<<<(int)(welems / 8 / 256), 256, 0, stream>>>(Wk, wkb, (int)welems);
    cvt_bf16<<<(int)(welems / 8 / 256), 256, 0, stream>>>(Wv, wvb, (int)welems);
    cvt_bf16<<<(int)(welems / 8 / 256), 256, 0, stream>>>(Wo, wob, (int)welems);

    dim3 gridP(MTOT / 128, D_MODEL / 64);
    gemm_bf16<1><<<gridP, 256, 0, stream>>>(xb, wqb, bq, q,  nullptr);
    gemm_bf16<1><<<gridP, 256, 0, stream>>>(xb, wkb, bk, k,  nullptr);
    gemm_bf16<2><<<gridP, 256, 0, stream>>>(xb, wvb, bv, vT, nullptr);
    flash_attn<<<BATCH * NHEADS * (SEQ / 64), 256, 0, stream>>>(q, k, vT, ao);
    gemm_bf16<0><<<gridP, 256, 0, stream>>>(ao, wob, bo, nullptr, out);
}
